// ScoreDecoder_17394617549346
// MI455X (gfx1250) — matrix operand order repaired and checked by reading, not compiled, not run on hardware
//
#include <hip/hip_runtime.h>
#include <hip/hip_bf16.h>
#include <math.h>

typedef __attribute__((ext_vector_type(16))) _Float16 v16h;
typedef __attribute__((ext_vector_type(8)))  float    v8f;

#define B_ 16
#define M_ 16
#define N_ 32
#define T_ 80
#define TCH 8   // timesteps per attr batch (256 threads = TCH * N_)

// ---- B-fragment loaders straight from global f32 weights (one-time, per wave) ----
// B fragment: 32(K) x 16(N); lane holds column N=lane&15; lanes 0-15 K=0..7,16..23,
// lanes 16-31 K=8..15,24..31 (CDNA5 ISA 7.12.2).
__device__ __forceinline__ v16h load_B_w1(const float* __restrict__ w,
                                          const float* __restrict__ bias,
                                          int colbase, int lane) {
  const int n = lane & 15, hi = lane >> 4;
  v16h b;
#pragma unroll
  for (int e = 0; e < 16; ++e) {
    const int v = e >> 1, j = e & 1;
    const int k = ((2 * v + j) & 7) + 8 * hi + 16 * (e >> 3);
    float val = 0.f;
    if (k < 10)       val = w[k * 64 + colbase + n];
    else if (k == 10) val = bias[colbase + n];  // bias row; A column k=10 is constant 1.0
    b[e] = (_Float16)val;
  }
  return b;
}

__device__ __forceinline__ v16h load_B_w2(const float* __restrict__ w, int colbase, int kbase, int lane) {
  const int n = lane & 15, hi = lane >> 4;
  v16h b;
#pragma unroll
  for (int e = 0; e < 16; ++e) {
    const int v = e >> 1, j = e & 1;
    const int k = ((2 * v + j) & 7) + 8 * hi + 16 * (e >> 3);
    b[e] = (_Float16)w[(kbase + k) * 256 + colbase + n];
  }
  return b;
}

__device__ __forceinline__ float block_sum(float v, float* buf, int tid) {
  buf[tid] = v;
  __syncthreads();
  for (int s = 128; s > 0; s >>= 1) {
    if (tid < s) buf[tid] += buf[tid + s];
    __syncthreads();
  }
  float r = buf[0];
  __syncthreads();
  return r;
}

// ---------------- weight-decoder head: batch-invariant (enc_in = ones) ----------------
__global__ __launch_bounds__(256)
void weights_kernel(const float* __restrict__ wd_w1, const float* __restrict__ wd_b1,
                    const float* __restrict__ wd_w2, const float* __restrict__ wd_b2,
                    float* __restrict__ out, float* __restrict__ ws) {
  __shared__ float h[64];
  __shared__ float w8[8];
  const int tid = threadIdx.x;
  if (tid < 64) {
    float s = wd_b1[tid];
    for (int k = 0; k < 256; ++k) s += wd_w1[k * 64 + tid];  // ones @ W1 = column sums
    h[tid] = (s > 0.f) ? s : (expf(s) - 1.f);                // elu
  }
  __syncthreads();
  if (tid < 8) {
    float o = wd_b2[tid];
    for (int j = 0; j < 64; ++j) o += h[j] * wd_w2[j * 8 + tid];
    float sp = (o > 20.f) ? o : log1pf(expf(o));             // softplus
    w8[tid] = sp;
    ws[tid] = sp;
  }
  __syncthreads();
  if (tid < B_ * 8) out[B_ * M_ + tid] = w8[tid & 7];
}

// ---------------- fused score kernel: one block per (b, m) ----------------
__global__ __launch_bounds__(256)
void score_main(const float* __restrict__ ego_traj,
                const float* __restrict__ agents_traj,
                const float* __restrict__ agents_states,
                const float* __restrict__ enc_w1, const float* __restrict__ enc_b1,
                const float* __restrict__ enc_w2, const float* __restrict__ enc_b2,
                const float* __restrict__ dec_w1, const float* __restrict__ dec_b1,
                const float* __restrict__ dec_w2, const float* __restrict__ dec_b2,
                const float* __restrict__ wvec,
                float* __restrict__ out_scores) {
  __shared__ alignas(16) float sAgt[N_ * T_ * 3];            // staged agents_traj (30 KB)
  __shared__ alignas(32) _Float16 fragA[TCH][2][32][16];     // MLP1 A fragments, 8 timesteps
  __shared__ alignas(32) _Float16 fragH[2][2][2][32][16];    // MLP2 A fragments, double buffer
  __shared__ float sEgo[T_][6];                              // ex, ey, eyaw, esp, evx, evy
  __shared__ float sFeat[256];
  __shared__ float sRed[256];
  __shared__ float sMask[32];
  __shared__ float sHdec[64];
  __shared__ float sInter[4];

  const int tid = threadIdx.x;
  const int blk = blockIdx.x;
  const int b = blk / M_, m = blk % M_;
  const float* ego = ego_traj + (size_t)(b * M_ + m) * T_ * 6;
  const float* agt = agents_traj + (size_t)(b * M_ + m) * N_ * T_ * 3;

  // ---- stage agents into LDS (coalesced), ego table, zero fragment pads ----
  {
    const float4* g4 = (const float4*)agt;
    float4* l4 = (float4*)sAgt;
    for (int i = tid; i < (N_ * T_ * 3) / 4; i += 256) l4[i] = g4[i];
  }
  for (int i = tid; i < TCH * 2 * 32 * 16; i += 256) ((_Float16*)fragA)[i] = (_Float16)0.f;
  if (tid < T_) {
    const float ex = ego[tid * 6 + 0], ey = ego[tid * 6 + 1];
    const float eyaw = ego[tid * 6 + 2], esp = ego[tid * 6 + 3];
    sEgo[tid][0] = ex; sEgo[tid][1] = ey; sEgo[tid][2] = eyaw; sEgo[tid][3] = esp;
    sEgo[tid][4] = esp * cosf(eyaw); sEgo[tid][5] = esp * sinf(eyaw);
  }
  // warm decoder-head weights into L2 for the tail (global_prefetch_b8)
  __builtin_prefetch(dec_w1 + tid * 64, 0, 1);
  __builtin_prefetch(dec_w1 + tid * 64 + 32, 0, 1);
  __syncthreads();

  // t-invariant attr features (agent states, k=5..9) + constant-1 bias column (k=10),
  // replicated into all TCH chunk slots.
  if (tid < N_) {
    const int n = tid;
    const float* stp = agents_states + (size_t)(b * N_ + n) * 11;
    float s = 0.f;
    for (int c = 0; c < 11; ++c) s += stp[c];
    const float mk = (s != 0.f) ? 1.f : 0.f;
    sMask[n] = mk;
    const int tl = n >> 4, mr = n & 15;
#pragma unroll
    for (int sl = 0; sl < TCH; ++sl) {
      _Float16* lo = fragA[sl][tl][mr];
      lo[5] = (_Float16)(stp[6] * mk);          // k=5 -> lane-lo e=5
      lo[6] = (_Float16)(stp[7] * mk);          // k=6
      lo[7] = (_Float16)(stp[8] * mk);          // k=7
      _Float16* hiR = fragA[sl][tl][mr + 16];
      hiR[0] = (_Float16)(stp[9] * mk);         // k=8 -> lane-hi e=0
      hiR[1] = (_Float16)(stp[10] * mk);        // k=9 -> lane-hi e=1
      hiR[2] = (_Float16)1.0f;                  // k=10 -> constant 1 (bias row in W1)
    }
  }
  __syncthreads();

  // ---- hard features ----
  float v0 = 0.f, v1 = 0.f, v2 = 0.f, v3 = 0.f;
  if (tid < T_) {
    const int t = tid;
    const float sp = ego[t * 6 + 3];
    const float ac = ego[t * 6 + 4];
    const int t1 = (t == 0) ? 1 : t;
    const float jk = (ego[t1 * 6 + 4] - ego[(t1 - 1) * 6 + 4]) * 10.0f;
    const float cv = ego[t * 6 + 5];
    v0 = sp; v1 = fabsf(ac); v2 = fabsf(jk); v3 = fabsf(sp * sp * cv);
  }
  const float ms = block_sum(v0, sRed, tid) * (1.f / T_);
  const float ma = block_sum(v1, sRed, tid) * (1.f / T_);
  const float mj = block_sum(v2, sRed, tid) * (1.f / T_);
  const float ml = block_sum(v3, sRed, tid) * (1.f / T_);
  const float hard0 = -fminf(fmaxf(ms, 0.f), 15.f) / 15.f;
  const float hard1 =  fminf(fmaxf(ma, 0.f), 4.f) / 4.f;
  const float hard2 =  fminf(fmaxf(mj, 0.f), 6.f) / 6.f;
  const float hard3 =  fminf(fmaxf(ml, 0.f), 5.f) / 5.f;

  // ---- collision term (LDS-staged agents) and ego mask ----
  float cacc = 0.f;
  for (int p = tid; p < N_ * T_; p += 256) {
    const int n = p / T_, t = p - n * T_;
    const float dx = sEgo[t][0] - sAgt[(n * T_ + t) * 3 + 0];
    const float dy = sEgo[t][1] - sAgt[(n * T_ + t) * 3 + 1];
    cacc += expf(-0.2f * (dx * dx + dy * dy)) * sMask[n];
  }
  const float collv = block_sum(cacc, sRed, tid);
  float es = 0.f;
  for (int p = tid; p < T_ * 6; p += 256) es += ego[p];
  const float egos = block_sum(es, sRed, tid);

  // ---- hoist loop-invariant WMMA operands into registers ----
  const int lane = tid & 31, wid = tid >> 5;
  const int nn = lane & 15;
  const int rt1 = wid >> 2, ct1 = wid & 3;                  // MLP1: 2 row x 4 col tiles
  const v16h bW1 = load_B_w1(enc_w1, enc_b1, ct1 * 16, lane);
  v16h bW2[2][2];
#pragma unroll
  for (int q = 0; q < 2; ++q)
#pragma unroll
    for (int kc = 0; kc < 2; ++kc)
      bW2[q][kc] = load_B_w2(enc_w2, (2 * wid + q) * 16, kc * 32, lane);
  const float bias2_0 = enc_b2[(2 * wid) * 16 + nn];
  const float bias2_1 = enc_b2[(2 * wid + 1) * 16 + nn];

  // MLP1 -> fragH scatter destination (lane-invariant except row index), per buffer
  const int cidx = ct1 * 16 + nn;
  const int kcw = cidx >> 5;
  const int kl = cidx & 31;
  int hik, ek;
  if (kl < 8)       { hik = 0; ek = kl; }
  else if (kl < 16) { hik = 1; ek = kl - 8; }
  else if (kl < 24) { hik = 0; ek = kl - 8; }
  else              { hik = 1; ek = kl - 16; }
  _Float16* h1dst[2] = { &fragH[0][rt1][kcw][8 * (lane >> 4) + 16 * hik][ek],
                         &fragH[1][rt1][kcw][8 * (lane >> 4) + 16 * hik][ek] };

  float facc0 = 0.f, facc1 = 0.f;

  auto mlp1 = [&](int ti) {
    v16h a = *(const v16h*)&fragA[ti][rt1][lane][0];        // 32B contiguous per lane
    v8f c = {};                                             // bias folded into K=10 row
    c = __builtin_amdgcn_wmma_f32_16x16x32_f16(false, a, false, bW1, (short)0, c, false, false);
    _Float16* dst = h1dst[ti & 1];
#pragma unroll
    for (int r = 0; r < 8; ++r) dst[r * 16] = (_Float16)fmaxf(c[r], 0.f);
  };

  auto mlp2 = [&](int ti) {
    const int bufi = ti & 1;
    v16h a00 = *(const v16h*)&fragH[bufi][0][0][lane][0];
    v16h a01 = *(const v16h*)&fragH[bufi][0][1][lane][0];
    v16h a10 = *(const v16h*)&fragH[bufi][1][0][lane][0];
    v16h a11 = *(const v16h*)&fragH[bufi][1][1][lane][0];
#pragma unroll
    for (int q = 0; q < 2; ++q) {
      const float bias = (q == 0) ? bias2_0 : bias2_1;
      v8f c0, c1;
#pragma unroll
      for (int r = 0; r < 8; ++r) { c0[r] = bias; c1[r] = bias; }
      c0 = __builtin_amdgcn_wmma_f32_16x16x32_f16(false, a00, false, bW2[q][0], (short)0, c0, false, false);
      c0 = __builtin_amdgcn_wmma_f32_16x16x32_f16(false, a01, false, bW2[q][1], (short)0, c0, false, false);
      c1 = __builtin_amdgcn_wmma_f32_16x16x32_f16(false, a10, false, bW2[q][0], (short)0, c1, false, false);
      c1 = __builtin_amdgcn_wmma_f32_16x16x32_f16(false, a11, false, bW2[q][1], (short)0, c1, false, false);
      float mx = -3.402823466e38f;
#pragma unroll
      for (int r = 0; r < 8; ++r) mx = fmaxf(mx, fmaxf(c0[r], c1[r]));
      mx = fmaxf(mx, __shfl_xor(mx, 16, 32));               // max over all 32 agents
      if (q == 0) facc0 += mx; else facc1 += mx;
    }
  };

  // ---- main loop: attr batched per 8 timesteps; MLP1/MLP2 software-pipelined ----
  for (int tc = 0; tc < T_; tc += TCH) {
    {
      // all 256 threads: (ti = tid/32) x (agent n = tid%32)
      const int ti = tid >> 5;
      const int n = tid & 31;
      const int t = tc + ti;
      const float ex = sEgo[t][0], ey = sEgo[t][1], eyaw = sEgo[t][2];
      const float evx = sEgo[t][4], evy = sEgo[t][5];
      const float* ap = &sAgt[(n * T_ + t) * 3];
      const float ax = ap[0], ay = ap[1], ayaw = ap[2];
      const float dyaw = ayaw - eyaw;
      const float sry = sinf(dyaw), cry = cosf(dyaw);
      const float ry = atan2f(sry, cry);                    // cos/sin(ry)==cry/sry
      const int t1 = (t == 0) ? 1 : t;
      const float* ap1 = &sAgt[(n * T_ + t1) * 3];
      const float avx = (ap1[0] - ap1[-3]) * 10.f;
      const float avy = (ap1[1] - ap1[-2]) * 10.f;
      const float mk = sMask[n];
      _Float16* row = fragA[ti][n >> 4][n & 15];
      row[0] = (_Float16)((ax - ex) * cry * mk);
      row[1] = (_Float16)((ay - ey) * sry * mk);
      row[2] = (_Float16)(ry * mk);
      row[3] = (_Float16)((avx - evx) * cry * mk);
      row[4] = (_Float16)((avy - evy) * sry * mk);
    }
    __syncthreads();

    mlp1(0);                         // prologue fills buffer 0
#pragma unroll 1
    for (int ti = 0; ti < TCH; ++ti) {
      __syncthreads();
      if (ti < TCH - 1) mlp1(ti + 1);  // fill buf[(ti+1)&1]
      mlp2(ti);                        // consume buf[ti&1]
    }
  }

  // feat = mean over T of max over N
  if (lane < 16) {
    sFeat[(2 * wid) * 16 + lane]     = facc0 * (1.f / T_);
    sFeat[(2 * wid + 1) * 16 + lane] = facc1 * (1.f / T_);
  }
  __syncthreads();

  // ---- decoder head ----
  if (tid < 64) {
    float h = dec_b1[tid];
    for (int k = 0; k < 256; ++k) h += sFeat[k] * dec_w1[k * 64 + tid];
    sHdec[tid] = (h > 0.f) ? h : (expf(h) - 1.f);  // elu
  }
  __syncthreads();
  if (tid < 4) {
    float o = dec_b2[tid];
    for (int j = 0; j < 64; ++j) o += sHdec[j] * dec_w2[j * 4 + tid];
    sInter[tid] = 1.f / (1.f + expf(-o));          // sigmoid
  }
  __syncthreads();

  if (tid == 0) {
    float sc = hard0 * wvec[0] + hard1 * wvec[1] + hard2 * wvec[2] + hard3 * wvec[3]
             + sInter[0] * wvec[4] + sInter[1] * wvec[5]
             + sInter[2] * wvec[6] + sInter[3] * wvec[7];
    float score = -sc - 10.f * collv;
    if (egos == 0.f) score = -__builtin_inff();
    out_scores[b * M_ + m] = score;
  }
}

extern "C" void kernel_launch(void* const* d_in, const int* in_sizes, int n_in,
                              void* d_out, int out_size, void* d_ws, size_t ws_size,
                              hipStream_t stream) {
  const float* ego_traj      = (const float*)d_in[0];
  const float* agents_traj   = (const float*)d_in[2];
  const float* agents_states = (const float*)d_in[3];
  const float* enc_w1 = (const float*)d_in[4];
  const float* enc_b1 = (const float*)d_in[5];
  const float* enc_w2 = (const float*)d_in[6];
  const float* enc_b2 = (const float*)d_in[7];
  const float* dec_w1 = (const float*)d_in[8];
  const float* dec_b1 = (const float*)d_in[9];
  const float* dec_w2 = (const float*)d_in[10];
  const float* dec_b2 = (const float*)d_in[11];
  const float* wd_w1  = (const float*)d_in[12];
  const float* wd_b1  = (const float*)d_in[13];
  const float* wd_w2  = (const float*)d_in[14];
  const float* wd_b2  = (const float*)d_in[15];
  float* out = (float*)d_out;
  float* ws  = (float*)d_ws;

  weights_kernel<<<1, 256, 0, stream>>>(wd_w1, wd_b1, wd_w2, wd_b2, out, ws);
  score_main<<<B_ * M_, 256, 0, stream>>>(ego_traj, agents_traj, agents_states,
                                          enc_w1, enc_b1, enc_w2, enc_b2,
                                          dec_w1, dec_b1, dec_w2, dec_b2,
                                          ws, out);
}
